// HetGATLayer_84404697301141
// MI455X (gfx1250) — matrix operand order repaired and checked by reading, not compiled, not run on hardware
//
#include <hip/hip_runtime.h>

typedef __attribute__((ext_vector_type(16))) __bf16 v16bf;
typedef __attribute__((ext_vector_type(8)))  __bf16 bf16x8;
typedef __attribute__((ext_vector_type(8)))  float  v8f;

#define H_    4
#define NPG_  10
#define NAG_  10
#define B_    1024
#define NP_   (B_*NPG_)
#define NA_   (B_*NAG_)
#define NS_   B_
#define DIN_  128
#define HID_  128

#define GEMM_BM   32      // M rows per block (shared A panel)
#define GEMM_BNW  32      // N cols per wave
#define GEMM_BN   128     // N cols per block (4 waves)
#define GEMM_KMAX 512

__device__ __forceinline__ float lrelu(float v){ return v > 0.f ? v : 0.2f*v; }

// ---------------- conversion kernels ----------------
__global__ void k_f32_to_bf16(const float* __restrict__ in, __bf16* __restrict__ out, int n){
  int i = blockIdx.x*blockDim.x + threadIdx.x;
  if (i < n) out[i] = (__bf16)in[i];
}

// in: K x N (row major)  ->  out: N x K (row major) bf16   (weight transpose)
__global__ void k_f32_to_bf16_T(const float* __restrict__ in, __bf16* __restrict__ out, int K, int N){
  int i = blockIdx.x*blockDim.x + threadIdx.x;
  if (i >= K*N) return;
  int n = i / K, k = i % K;
  out[(size_t)n*K + k] = (__bf16)in[(size_t)k*N + n];
}

// ---- WMMA fragment loaders (CDNA5 16-bit A 16x32 / B 32x16 lane layouts) ----
__device__ __forceinline__ v16bf load_fragA(const __bf16* __restrict__ row, int kk, int half){
  // e<8:  k = kk + half*8 + e ;  e>=8: k = kk + 16 + half*8 + (e-8)
  bf16x8 lo = *(const bf16x8*)(row + kk + half*8);
  bf16x8 hi = *(const bf16x8*)(row + kk + 16 + half*8);
  v16bf r;
#pragma unroll
  for (int e = 0; e < 8; ++e) { r[e] = lo[e]; r[e+8] = hi[e]; }
  return r;
}
__device__ __forceinline__ v16bf load_fragB(const __bf16* __restrict__ row, int kk, int half){
  // e<8:  k = kk + half*8 + e ;  e>=8: k = kk + 16 + half*8 + (e-8)
  bf16x8 lo = *(const bf16x8*)(row + kk + half*8);
  bf16x8 hi = *(const bf16x8*)(row + kk + 16 + half*8);
  v16bf r;
#pragma unroll
  for (int e = 0; e < 8; ++e) { r[e] = lo[e]; r[e+8] = hi[e]; }
  return r;
}

// ---------------- WMMA GEMM:  C[M,N] = A[M,K](bf16) @ Bt[N,K]^T(bf16) + bias ----------------
// Block = 4 waves; block owns 32 M-rows (A panel async-staged to LDS once) x 128 N-cols.
// Each wave computes a 32x32 tile: 2 A-frags (from LDS) x 2 B-frags (global) -> 4 WMMAs / K-step.
__global__ void __launch_bounds__(128)
k_gemm_wmma_bf16(const __bf16* __restrict__ A,
                 const __bf16* __restrict__ Bt,
                 const float*  __restrict__ bias,
                 float* __restrict__ C,
                 int M, int N, int K){
  __shared__ __align__(16) __bf16 ldsA[GEMM_BM * GEMM_KMAX];

  const int tid  = threadIdx.x;
  const int lane = tid & 31;
  const int w    = tid >> 5;
  const int blocksN = (N + GEMM_BN - 1) / GEMM_BN;
  const int bM = blockIdx.x / blocksN;
  const int bN = blockIdx.x % blocksN;

  // ---- async-stage the 32 x K A panel (32 consecutive rows = one contiguous block) ----
  {
    const unsigned chunks = (unsigned)(GEMM_BM * K) / 8u;            // 16-byte chunks
    const unsigned long long gbase =
        (unsigned long long)(size_t)(A + (size_t)bM * GEMM_BM * K);
    const unsigned ldsBase = (unsigned)(size_t)&ldsA[0];
    for (unsigned c = tid; c < chunks; c += 128u) {
      unsigned loff = ldsBase + c * 16u;
      unsigned long long gaddr = gbase + (unsigned long long)c * 16ull;
      asm volatile("global_load_async_to_lds_b128 %0, %1, off"
                   :: "v"(loff), "v"(gaddr) : "memory");
    }
  }
#if __has_builtin(__builtin_amdgcn_s_wait_asynccnt)
  __builtin_amdgcn_s_wait_asynccnt(0);
#else
  asm volatile("s_wait_asynccnt 0x0" ::: "memory");
#endif
  __syncthreads();

  const int half = lane >> 4;
  const int l16  = lane & 15;
  const int n0   = bN * GEMM_BN + w * GEMM_BNW;     // wave-uniform
  if (n0 < N) {
    const __bf16* a0Row = &ldsA[(size_t)l16 * K];
    const __bf16* a1Row = &ldsA[(size_t)(16 + l16) * K];
    const __bf16* b0Row = Bt + (size_t)(n0 + l16) * K;
    const __bf16* b1Row = Bt + (size_t)(n0 + 16 + l16) * K;

    v8f acc00 = {}, acc01 = {}, acc10 = {}, acc11 = {};
    for (int kk = 0; kk < K; kk += 32) {
      if (kk + 32 < K) {
        __builtin_prefetch(b0Row + kk + 32, 0, 3);
        __builtin_prefetch(b1Row + kk + 32, 0, 3);
      }
      v16bf a0 = load_fragA(a0Row, kk, half);       // ds_load path
      v16bf a1 = load_fragA(a1Row, kk, half);
      v16bf b0 = load_fragB(b0Row, kk, half);       // global path
      v16bf b1 = load_fragB(b1Row, kk, half);
      acc00 = __builtin_amdgcn_wmma_f32_16x16x32_bf16(false, a0, false, b0, (short)0, acc00, false, false);
      acc01 = __builtin_amdgcn_wmma_f32_16x16x32_bf16(false, a0, false, b1, (short)0, acc01, false, false);
      acc10 = __builtin_amdgcn_wmma_f32_16x16x32_bf16(false, a1, false, b0, (short)0, acc10, false, false);
      acc11 = __builtin_amdgcn_wmma_f32_16x16x32_bf16(false, a1, false, b1, (short)0, acc11, false, false);
    }
    const int ng0 = n0 + l16, ng1 = n0 + 16 + l16;
    const float bn0 = bias ? bias[ng0] : 0.0f;
    const float bn1 = bias ? bias[ng1] : 0.0f;
#pragma unroll
    for (int r = 0; r < 8; ++r) {
      int m0 = bM*GEMM_BM + r + half*8;             // C/D layout: vgpr r, lane-half -> M+8
      int m1 = m0 + 16;
      C[(size_t)m0*N + ng0] = acc00[r] + bn0;
      C[(size_t)m0*N + ng1] = acc01[r] + bn1;
      C[(size_t)m1*N + ng0] = acc10[r] + bn0;
      C[(size_t)m1*N + ng1] = acc11[r] + bn1;
    }
  }
}

// tiny K=4 projection (layer-1 state):  out[M,N] = X[M,4] @ W[4,N] + b
__global__ void k_gemm_k4(const float* __restrict__ X, const float* __restrict__ W,
                          const float* __restrict__ b, float* __restrict__ out,
                          int M, int N){
  int i = blockIdx.x*blockDim.x + threadIdx.x;
  if (i >= M*N) return;
  int m = i / N, n = i % N;
  float acc = b[n];
#pragma unroll
  for (int k = 0; k < 4; ++k) acc += X[m*4+k]*W[k*N+n];
  out[i] = acc;
}

// ---------------- attention scores: s[n,h] = dot(z[n,h,:], a[h,:]) ----------------
__global__ void k_attn_score(const float* __restrict__ z, const float* __restrict__ a,
                             float* __restrict__ s, int nNodes, int D){
  int i = blockIdx.x*blockDim.x + threadIdx.x;
  if (i >= nNodes*H_) return;
  int h = i & (H_-1);
  const float* zp = z + (size_t)i * D;    // z is [n,H,D] flattened; i = n*H+h
  const float* ap = a + (size_t)h * D;
  float acc = 0.f;
  for (int d = 0; d < D; ++d) acc += zp[d]*ap[d];
  s[i] = acc;
}

// 10-way leaky-relu softmax in registers (fixed segment structure)
__device__ __forceinline__ void softmax10(const float* __restrict__ ssrc, float sdst,
                                          int sb, int h, float* alpha){
  float e[10]; float mx = -3.4e38f;
#pragma unroll
  for (int j = 0; j < 10; ++j) {
    float v = lrelu(ssrc[(sb+j)*H_ + h] + sdst);
    e[j] = v; mx = fmaxf(mx, v);
  }
  float den = 0.f;
#pragma unroll
  for (int j = 0; j < 10; ++j) { e[j] = __expf(e[j]-mx); den += e[j]; }
  float inv = 1.0f/den;
#pragma unroll
  for (int j = 0; j < 10; ++j) alpha[j] = e[j]*inv;
}

// ---------------- layer-1 aggregation for P/A destinations ----------------
__global__ void k_agg1_PA(const float* __restrict__ z1, const float* __restrict__ s1s, const float* __restrict__ s1d,
                          const float* __restrict__ z2, const float* __restrict__ s2s, const float* __restrict__ s2d,
                          __bf16* __restrict__ h1out, int nDst){
  int lane = threadIdx.x & 31;
  int wid  = (blockIdx.x * (blockDim.x >> 5)) + (threadIdx.x >> 5);
  if (wid >= nDst*H_) return;
  int n = wid >> 2, h = wid & 3;
  int sb = (n / NPG_) * NPG_;
  float a0=0,a1=0,a2=0,a3=0;
  float alpha[10];
  softmax10(s1s, s1d[n*H_+h], sb, h, alpha);
#pragma unroll
  for (int j = 0; j < 10; ++j) {
    const float* zp = z1 + ((size_t)(sb+j)*H_ + h)*HID_;
    float al = alpha[j];
    a0 += al*zp[lane]; a1 += al*zp[lane+32]; a2 += al*zp[lane+64]; a3 += al*zp[lane+96];
  }
  softmax10(s2s, s2d[n*H_+h], sb, h, alpha);
#pragma unroll
  for (int j = 0; j < 10; ++j) {
    const float* zp = z2 + ((size_t)(sb+j)*H_ + h)*HID_;
    float al = alpha[j];
    a0 += al*zp[lane]; a1 += al*zp[lane+32]; a2 += al*zp[lane+64]; a3 += al*zp[lane+96];
  }
  size_t o = (size_t)n*(H_*HID_) + (size_t)h*HID_ + lane;
  h1out[o]    = (__bf16)fmaxf(a0, 0.f);
  h1out[o+32] = (__bf16)fmaxf(a1, 0.f);
  h1out[o+64] = (__bf16)fmaxf(a2, 0.f);
  h1out[o+96] = (__bf16)fmaxf(a3, 0.f);
}

// layer-1 aggregation for S destinations (p2s + a2s + self 'in' relation)
__global__ void k_agg1_S(const float* __restrict__ zps, const float* __restrict__ sps, const float* __restrict__ spd,
                         const float* __restrict__ zas, const float* __restrict__ sas, const float* __restrict__ sad,
                         const float* __restrict__ Ws, __bf16* __restrict__ h1out, int nS){
  int lane = threadIdx.x & 31;
  int wid  = (blockIdx.x * (blockDim.x >> 5)) + (threadIdx.x >> 5);
  if (wid >= nS*H_) return;
  int n = wid >> 2, h = wid & 3;
  int sb = n * NPG_;
  float a0=0,a1=0,a2=0,a3=0;
  float alpha[10];
  softmax10(sps, spd[n*H_+h], sb, h, alpha);
#pragma unroll
  for (int j = 0; j < 10; ++j) {
    const float* zp = zps + ((size_t)(sb+j)*H_ + h)*HID_;
    float al = alpha[j];
    a0 += al*zp[lane]; a1 += al*zp[lane+32]; a2 += al*zp[lane+64]; a3 += al*zp[lane+96];
  }
  softmax10(sas, sad[n*H_+h], sb, h, alpha);
#pragma unroll
  for (int j = 0; j < 10; ++j) {
    const float* zp = zas + ((size_t)(sb+j)*H_ + h)*HID_;
    float al = alpha[j];
    a0 += al*zp[lane]; a1 += al*zp[lane+32]; a2 += al*zp[lane+64]; a3 += al*zp[lane+96];
  }
  const float* wp = Ws + ((size_t)n*H_ + h)*HID_;  // self-loop: softmax over 1 edge -> alpha=1
  a0 += wp[lane]; a1 += wp[lane+32]; a2 += wp[lane+64]; a3 += wp[lane+96];
  size_t o = (size_t)n*(H_*HID_) + (size_t)h*HID_ + lane;
  h1out[o]    = (__bf16)fmaxf(a0, 0.f);
  h1out[o+32] = (__bf16)fmaxf(a1, 0.f);
  h1out[o+64] = (__bf16)fmaxf(a2, 0.f);
  h1out[o+96] = (__bf16)fmaxf(a3, 0.f);
}

// ---------------- layer-2 aggregation, P/A destinations: head-average -> d_out ----------------
__global__ void k_agg2_PA(const float* __restrict__ z1, const float* __restrict__ s1s, const float* __restrict__ s1d,
                          const float* __restrict__ z2, const float* __restrict__ s2s, const float* __restrict__ s2d,
                          float* __restrict__ out, int nDst, int rowOff){
  int lane = threadIdx.x & 31;
  int wid  = (blockIdx.x * (blockDim.x >> 5)) + (threadIdx.x >> 5);
  if (wid >= nDst) return;
  int n = wid, g = n / NPG_, i = n % NPG_, sb = g * NPG_;
  float a0=0,a1=0,a2=0,a3=0;
  for (int h = 0; h < H_; ++h) {
    float alpha[10];
    softmax10(s1s, s1d[n*H_+h], sb, h, alpha);
#pragma unroll
    for (int j = 0; j < 10; ++j) {
      const float* zp = z1 + ((size_t)(sb+j)*H_ + h)*DIN_;
      float al = 0.25f*alpha[j];
      a0 += al*zp[lane]; a1 += al*zp[lane+32]; a2 += al*zp[lane+64]; a3 += al*zp[lane+96];
    }
    softmax10(s2s, s2d[n*H_+h], sb, h, alpha);
#pragma unroll
    for (int j = 0; j < 10; ++j) {
      const float* zp = z2 + ((size_t)(sb+j)*H_ + h)*DIN_;
      float al = 0.25f*alpha[j];
      a0 += al*zp[lane]; a1 += al*zp[lane+32]; a2 += al*zp[lane+64]; a3 += al*zp[lane+96];
    }
  }
  size_t o = ((size_t)(g*(NPG_+NAG_) + rowOff + i))*DIN_ + lane;
  out[o] = a0; out[o+32] = a1; out[o+64] = a2; out[o+96] = a3;
}

// layer-2 aggregation, S destinations (D=8): head-average -> qS section of d_out
__global__ void k_agg2_S(const float* __restrict__ zps, const float* __restrict__ sps, const float* __restrict__ spd,
                         const float* __restrict__ zas, const float* __restrict__ sas, const float* __restrict__ sad,
                         const float* __restrict__ W2s, float* __restrict__ out, int nS){
  int lane = threadIdx.x & 31;
  int wid  = (blockIdx.x * (blockDim.x >> 5)) + (threadIdx.x >> 5);
  if (wid >= nS) return;
  if (lane >= 8) return;
  int n = wid, sb = n * NPG_;
  float acc = 0.f;
  for (int h = 0; h < H_; ++h) {
    float alpha[10];
    softmax10(sps, spd[n*H_+h], sb, h, alpha);
#pragma unroll
    for (int j = 0; j < 10; ++j)
      acc += 0.25f*alpha[j]*zps[((size_t)(sb+j)*H_ + h)*8 + lane];
    softmax10(sas, sad[n*H_+h], sb, h, alpha);
#pragma unroll
    for (int j = 0; j < 10; ++j)
      acc += 0.25f*alpha[j]*zas[((size_t)(sb+j)*H_ + h)*8 + lane];
    acc += 0.25f*W2s[((size_t)n*H_ + h)*8 + lane];   // self-loop alpha=1
  }
  out[(size_t)n*8 + lane] = acc;
}

// =====================================================================================
extern "C" void kernel_launch(void* const* d_in, const int* in_sizes, int n_in,
                              void* d_out, int out_size, void* d_ws, size_t ws_size,
                              hipStream_t stream) {
  (void)in_sizes; (void)n_in; (void)out_size; (void)ws_size;
  const float* hP   = (const float*)d_in[0];
  const float* hA   = (const float*)d_in[1];
  const float* cent = (const float*)d_in[2];
  // params1 entries start at 17, params2 at 41 (dict insertion order)
  auto P1 = [&](int i){ return (const float*)d_in[17 + i]; };
  auto P2 = [&](int i){ return (const float*)d_in[41 + i]; };
  float* out = (float*)d_out;

  auto cdiv = [](long a, long b){ return (int)((a + b - 1) / b); };

  // ---- workspace arena ----
  char* base = (char*)d_ws;
  size_t off = 0;
  auto alloc = [&](size_t bytes)->void* {
    void* p = base + off;
    off += (bytes + 255) & ~(size_t)255;
    return p;
  };

  // persistent across phases
  __bf16* h1Pb = (__bf16*)alloc((size_t)NP_*512*2);
  __bf16* h1Ab = (__bf16*)alloc((size_t)NA_*512*2);
  __bf16* h1Sb = (__bf16*)alloc((size_t)NS_*512*2);
  size_t scratch0 = off;

  // ================= PHASE 1 =================
  __bf16* hPb  = (__bf16*)alloc((size_t)NP_*DIN_*2);
  __bf16* hAb  = (__bf16*)alloc((size_t)NA_*DIN_*2);
  __bf16* w1P  = (__bf16*)alloc((size_t)512*DIN_*2);
  __bf16* w1A  = (__bf16*)alloc((size_t)512*DIN_*2);
  __bf16* w1PS = (__bf16*)alloc((size_t)512*DIN_*2);
  __bf16* w1AS = (__bf16*)alloc((size_t)512*DIN_*2);
  float* Wp  = (float*)alloc((size_t)NP_*512*4);
  float* Wa  = (float*)alloc((size_t)NA_*512*4);
  float* Wps = (float*)alloc((size_t)NP_*512*4);
  float* Was = (float*)alloc((size_t)NA_*512*4);
  float* Ws  = (float*)alloc((size_t)NS_*512*4);
  float* s_p2p_s = (float*)alloc((size_t)NP_*H_*4);
  float* s_p2p_d = (float*)alloc((size_t)NP_*H_*4);
  float* s_p2a_s = (float*)alloc((size_t)NP_*H_*4);
  float* s_p2a_d = (float*)alloc((size_t)NA_*H_*4);
  float* s_a2p_s = (float*)alloc((size_t)NA_*H_*4);
  float* s_a2p_d = (float*)alloc((size_t)NP_*H_*4);
  float* s_a2a_s = (float*)alloc((size_t)NA_*H_*4);
  float* s_a2a_d = (float*)alloc((size_t)NA_*H_*4);
  float* s_p2s_s = (float*)alloc((size_t)NP_*H_*4);
  float* s_p2s_d = (float*)alloc((size_t)NS_*H_*4);
  float* s_a2s_s = (float*)alloc((size_t)NA_*H_*4);
  float* s_a2s_d = (float*)alloc((size_t)NS_*H_*4);

  // conversions
  k_f32_to_bf16<<<cdiv((long)NP_*DIN_,256),256,0,stream>>>(hP, hPb, NP_*DIN_);
  k_f32_to_bf16<<<cdiv((long)NA_*DIN_,256),256,0,stream>>>(hA, hAb, NA_*DIN_);
  k_f32_to_bf16_T<<<cdiv((long)DIN_*512,256),256,0,stream>>>(P1(0), w1P,  DIN_, 512);
  k_f32_to_bf16_T<<<cdiv((long)DIN_*512,256),256,0,stream>>>(P1(2), w1A,  DIN_, 512);
  k_f32_to_bf16_T<<<cdiv((long)DIN_*512,256),256,0,stream>>>(P1(6), w1PS, DIN_, 512);
  k_f32_to_bf16_T<<<cdiv((long)DIN_*512,256),256,0,stream>>>(P1(8), w1AS, DIN_, 512);

  // WMMA projections: grid = (M/32) * ceil(N/128), 128 threads (4 waves) / block
  auto gemmGrid = [&](int M, int N){ return (M/GEMM_BM) * cdiv(N, GEMM_BN); };
  k_gemm_wmma_bf16<<<gemmGrid(NP_,512),128,0,stream>>>(hPb, w1P,  P1(1), Wp,  NP_, 512, DIN_);
  k_gemm_wmma_bf16<<<gemmGrid(NA_,512),128,0,stream>>>(hAb, w1A,  P1(3), Wa,  NA_, 512, DIN_);
  k_gemm_wmma_bf16<<<gemmGrid(NP_,512),128,0,stream>>>(hPb, w1PS, P1(7), Wps, NP_, 512, DIN_);
  k_gemm_wmma_bf16<<<gemmGrid(NA_,512),128,0,stream>>>(hAb, w1AS, P1(9), Was, NA_, 512, DIN_);
  k_gemm_k4<<<cdiv((long)NS_*512,256),256,0,stream>>>(cent, P1(4), P1(5), Ws, NS_, 512);

  // attention scores
  auto scoreGrid = [&](int n){ return cdiv((long)n*H_, 256); };
  k_attn_score<<<scoreGrid(NP_),256,0,stream>>>(Wp,  P1(10), s_p2p_s, NP_, HID_);
  k_attn_score<<<scoreGrid(NP_),256,0,stream>>>(Wp,  P1(11), s_p2p_d, NP_, HID_);
  k_attn_score<<<scoreGrid(NP_),256,0,stream>>>(Wp,  P1(12), s_p2a_s, NP_, HID_);
  k_attn_score<<<scoreGrid(NA_),256,0,stream>>>(Wa,  P1(13), s_p2a_d, NA_, HID_);
  k_attn_score<<<scoreGrid(NA_),256,0,stream>>>(Wa,  P1(14), s_a2p_s, NA_, HID_);
  k_attn_score<<<scoreGrid(NP_),256,0,stream>>>(Wp,  P1(15), s_a2p_d, NP_, HID_);
  k_attn_score<<<scoreGrid(NA_),256,0,stream>>>(Wa,  P1(16), s_a2a_s, NA_, HID_);
  k_attn_score<<<scoreGrid(NA_),256,0,stream>>>(Wa,  P1(17), s_a2a_d, NA_, HID_);
  k_attn_score<<<scoreGrid(NP_),256,0,stream>>>(Wps, P1(18), s_p2s_s, NP_, HID_);
  k_attn_score<<<scoreGrid(NS_),256,0,stream>>>(Ws,  P1(19), s_p2s_d, NS_, HID_);
  k_attn_score<<<scoreGrid(NA_),256,0,stream>>>(Was, P1(20), s_a2s_s, NA_, HID_);
  k_attn_score<<<scoreGrid(NS_),256,0,stream>>>(Ws,  P1(21), s_a2s_d, NS_, HID_);

  // aggregation (one wave per (dst,head); 4 waves/block)
  k_agg1_PA<<<cdiv((long)NP_*H_,4),128,0,stream>>>(Wp, s_p2p_s, s_p2p_d, Wa, s_a2p_s, s_a2p_d, h1Pb, NP_);
  k_agg1_PA<<<cdiv((long)NA_*H_,4),128,0,stream>>>(Wp, s_p2a_s, s_p2a_d, Wa, s_a2a_s, s_a2a_d, h1Ab, NA_);
  k_agg1_S <<<cdiv((long)NS_*H_,4),128,0,stream>>>(Wps, s_p2s_s, s_p2s_d, Was, s_a2s_s, s_a2s_d, Ws, h1Sb, NS_);

  // ================= PHASE 2 (reuse scratch) =================
  off = scratch0;
  __bf16* w2P  = (__bf16*)alloc((size_t)512*512*2);
  __bf16* w2A  = (__bf16*)alloc((size_t)512*512*2);
  __bf16* w2S  = (__bf16*)alloc((size_t)32*512*2);
  __bf16* w2PS = (__bf16*)alloc((size_t)32*512*2);
  __bf16* w2AS = (__bf16*)alloc((size_t)32*512*2);
  float* W2p  = (float*)alloc((size_t)NP_*512*4);
  float* W2a  = (float*)alloc((size_t)NA_*512*4);
  float* W2s  = (float*)alloc((size_t)NS_*32*4);
  float* W2ps = (float*)alloc((size_t)NP_*32*4);
  float* W2as = (float*)alloc((size_t)NA_*32*4);
  float* t_p2p_s = (float*)alloc((size_t)NP_*H_*4);
  float* t_p2p_d = (float*)alloc((size_t)NP_*H_*4);
  float* t_p2a_s = (float*)alloc((size_t)NP_*H_*4);
  float* t_p2a_d = (float*)alloc((size_t)NA_*H_*4);
  float* t_a2p_s = (float*)alloc((size_t)NA_*H_*4);
  float* t_a2p_d = (float*)alloc((size_t)NP_*H_*4);
  float* t_a2a_s = (float*)alloc((size_t)NA_*H_*4);
  float* t_a2a_d = (float*)alloc((size_t)NA_*H_*4);
  float* t_p2s_s = (float*)alloc((size_t)NP_*H_*4);
  float* t_p2s_d = (float*)alloc((size_t)NS_*H_*4);
  float* t_a2s_s = (float*)alloc((size_t)NA_*H_*4);
  float* t_a2s_d = (float*)alloc((size_t)NS_*H_*4);

  k_f32_to_bf16_T<<<cdiv((long)512*512,256),256,0,stream>>>(P2(0), w2P,  512, 512);
  k_f32_to_bf16_T<<<cdiv((long)512*512,256),256,0,stream>>>(P2(2), w2A,  512, 512);
  k_f32_to_bf16_T<<<cdiv((long)512*32, 256),256,0,stream>>>(P2(4), w2S,  512, 32);
  k_f32_to_bf16_T<<<cdiv((long)512*32, 256),256,0,stream>>>(P2(6), w2PS, 512, 32);
  k_f32_to_bf16_T<<<cdiv((long)512*32, 256),256,0,stream>>>(P2(8), w2AS, 512, 32);

  k_gemm_wmma_bf16<<<gemmGrid(NP_,512),128,0,stream>>>(h1Pb, w2P,  P2(1), W2p,  NP_, 512, 512);
  k_gemm_wmma_bf16<<<gemmGrid(NA_,512),128,0,stream>>>(h1Ab, w2A,  P2(3), W2a,  NA_, 512, 512);
  k_gemm_wmma_bf16<<<gemmGrid(NS_,32), 128,0,stream>>>(h1Sb, w2S,  P2(5), W2s,  NS_, 32,  512);
  k_gemm_wmma_bf16<<<gemmGrid(NP_,32), 128,0,stream>>>(h1Pb, w2PS, P2(7), W2ps, NP_, 32,  512);
  k_gemm_wmma_bf16<<<gemmGrid(NA_,32), 128,0,stream>>>(h1Ab, w2AS, P2(9), W2as, NA_, 32,  512);

  k_attn_score<<<scoreGrid(NP_),256,0,stream>>>(W2p,  P2(10), t_p2p_s, NP_, DIN_);
  k_attn_score<<<scoreGrid(NP_),256,0,stream>>>(W2p,  P2(11), t_p2p_d, NP_, DIN_);
  k_attn_score<<<scoreGrid(NP_),256,0,stream>>>(W2p,  P2(12), t_p2a_s, NP_, DIN_);
  k_attn_score<<<scoreGrid(NA_),256,0,stream>>>(W2a,  P2(13), t_p2a_d, NA_, DIN_);
  k_attn_score<<<scoreGrid(NA_),256,0,stream>>>(W2a,  P2(14), t_a2p_s, NA_, DIN_);
  k_attn_score<<<scoreGrid(NP_),256,0,stream>>>(W2p,  P2(15), t_a2p_d, NP_, DIN_);
  k_attn_score<<<scoreGrid(NA_),256,0,stream>>>(W2a,  P2(16), t_a2a_s, NA_, DIN_);
  k_attn_score<<<scoreGrid(NA_),256,0,stream>>>(W2a,  P2(17), t_a2a_d, NA_, DIN_);
  k_attn_score<<<scoreGrid(NP_),256,0,stream>>>(W2ps, P2(18), t_p2s_s, NP_, 8);
  k_attn_score<<<scoreGrid(NS_),256,0,stream>>>(W2s,  P2(19), t_p2s_d, NS_, 8);
  k_attn_score<<<scoreGrid(NA_),256,0,stream>>>(W2as, P2(20), t_a2s_s, NA_, 8);
  k_attn_score<<<scoreGrid(NS_),256,0,stream>>>(W2s,  P2(21), t_a2s_d, NS_, 8);

  // final aggregation: head-average, scattered into output layout (B,20,128) ++ (B,8)
  k_agg2_PA<<<cdiv((long)NP_,4),128,0,stream>>>(W2p, t_p2p_s, t_p2p_d, W2a, t_a2p_s, t_a2p_d, out, NP_, 0);
  k_agg2_PA<<<cdiv((long)NA_,4),128,0,stream>>>(W2p, t_p2a_s, t_p2a_d, W2a, t_a2a_s, t_a2a_d, out, NA_, NPG_);
  k_agg2_S <<<cdiv((long)NS_,4),128,0,stream>>>(W2ps, t_p2s_s, t_p2s_d, W2as, t_a2s_s, t_a2s_d, W2s,
                                                out + (size_t)B_*(NPG_+NAG_)*DIN_, NS_);
}
